// DiagnosticAttention_81905026334771
// MI455X (gfx1250) — matrix operand order repaired and checked by reading, not compiled, not run on hardware
//
#include <hip/hip_runtime.h>
#include <hip/hip_bf16.h>

// ---------------------------------------------------------------------------
// DiagnosticAttention for MI455X (gfx1250): bf16 WMMA GEMMs + flash attention
// B=2, L=2048, H=1024, NH=16, HD=64
// ---------------------------------------------------------------------------

typedef unsigned short u16;
typedef unsigned int   u32;
typedef unsigned long long u64;
typedef __attribute__((ext_vector_type(16))) __bf16 v16bf;
typedef __attribute__((ext_vector_type(8)))  float  v8f;

union FragAB { u32 u[8]; uint4 q[2]; v16bf v; };

#define LOG2E 1.44269504088896340736f

// hardware packed f32 -> bf16 (RNE), one VOP3 instruction
__device__ __forceinline__ u32 cvt_pk_bf16(float lo, float hi) {
    u32 r;
    asm("v_cvt_pk_bf16_f32 %0, %1, %2" : "=v"(r) : "v"(lo), "v"(hi));
    return r;
}

#define WMMA_BF16(A, B, C) \
    __builtin_amdgcn_wmma_f32_16x16x32_bf16(false, (A), false, (B), (short)0, (C), false, false)

// ---- cross-lane helpers (butterfly within each 16-lane group, VALU pipe) ----
__device__ __forceinline__ u32 plx16(u32 v, u32 a, u32 b) {
    return __builtin_amdgcn_permlane16(v, v, a, b, false, false);
}
// packed bf16 max: single VOP3P instruction (IEEE maximumNumber)
__device__ __forceinline__ u32 bfmax2(u32 a, u32 b) {
    u32 r;
    asm("v_pk_max_num_bf16 %0, %1, %2" : "=v"(r) : "v"(a), "v"(b));
    return r;
}
// max over the 16-lane group, two rows packed per register (bf16)
__device__ __forceinline__ u32 redmax16_bf2(u32 v) {
    v = bfmax2(v, plx16(v, 0x67452301u, 0xEFCDAB89u));  // xor 1
    v = bfmax2(v, plx16(v, 0x45670123u, 0xCDEF89ABu));  // xor 2
    v = bfmax2(v, plx16(v, 0x01234567u, 0x89ABCDEFu));  // xor 4
    v = bfmax2(v, plx16(v, 0xFEDCBA98u, 0x76543210u));  // xor 8
    return v;
}
// f32 sum over the 16-lane group (used once per wave at the end)
__device__ __forceinline__ float redadd16_f32(float v) {
    v += __uint_as_float(plx16(__float_as_uint(v), 0x67452301u, 0xEFCDAB89u));
    v += __uint_as_float(plx16(__float_as_uint(v), 0x45670123u, 0xCDEF89ABu));
    v += __uint_as_float(plx16(__float_as_uint(v), 0x01234567u, 0x89ABCDEFu));
    v += __uint_as_float(plx16(__float_as_uint(v), 0xFEDCBA98u, 0x76543210u));
    return v;
}

// ---- CDNA5 async copy: global -> LDS, 16B per lane, tracked by ASYNCcnt ----
__device__ __forceinline__ void async_b128(u32 lds_off, const void* gaddr) {
    asm volatile("global_load_async_to_lds_b128 %0, %1, off"
                 :: "v"(lds_off), "v"((u64)(size_t)gaddr) : "memory");
}
__device__ __forceinline__ void async_b128_o16(u32 lds_off, const void* gaddr) {
    // INST_OFFSET applies to BOTH the LDS and global addresses (ISA 08 §4.4)
    asm volatile("global_load_async_to_lds_b128 %0, %1, off offset:16"
                 :: "v"(lds_off), "v"((u64)(size_t)gaddr) : "memory");
}
__device__ __forceinline__ void wait_async0() {
    asm volatile("s_wait_asynccnt 0x0" ::: "memory");
}

// ---------------------------------------------------------------------------
// Prep kernels
// ---------------------------------------------------------------------------

// Transpose+convert Wq|Wk|Wv -> wt_qkv (3072,1024) and Wo -> wt_o (1024,1024).
// One k-pair per thread: two strided f32 reads, one packed-bf16 u32 store.
__global__ __launch_bounds__(256) void prep_w_kernel(
    const float* __restrict__ Wq, const float* __restrict__ Wk,
    const float* __restrict__ Wv, const float* __restrict__ Wo,
    u16* __restrict__ wt_qkv, u16* __restrict__ wt_o)
{
    int p = blockIdx.x * 256 + threadIdx.x;            // 0 .. 2M-1 (pairs)
    int idx = p * 2;
    if (idx < 3 * 1024 * 1024) {
        int n = idx >> 10, k = idx & 1023;             // k even, k+1 same row
        int which = n >> 10, nn = n & 1023;
        const float* W = (which == 0) ? Wq : (which == 1) ? Wk : Wv;
        u32 w = cvt_pk_bf16(W[k * 1024 + nn], W[(k + 1) * 1024 + nn]);
        *(u32*)(wt_qkv + idx) = w;
    } else {
        int i2 = idx - 3 * 1024 * 1024;
        int n = i2 >> 10, k = i2 & 1023;
        u32 w = cvt_pk_bf16(Wo[k * 1024 + n], Wo[(k + 1) * 1024 + n]);
        *(u32*)(wt_o + i2) = w;
    }
}

__global__ __launch_bounds__(256) void cvt_x_kernel(
    const float* __restrict__ x, u16* __restrict__ xb)
{
    int i = (blockIdx.x * 256 + threadIdx.x) * 2;      // 2M pairs
    float2 f = *(const float2*)(x + i);
    *(u32*)(xb + i) = cvt_pk_bf16(f.x, f.y);
}

// addterm[b,key,h] = log2e * (err*sigmoid(x·Wg+bg) + attention_mask + diag_bias)
// (pre-scaled by log2e so attention can use raw v_exp_f32 / exp2)
__global__ __launch_bounds__(256) void gate_kernel(
    const float* __restrict__ x, const float* __restrict__ Wg,
    const float* __restrict__ bg, const int* __restrict__ err,
    const float* __restrict__ amask, const float* __restrict__ diag,
    float* __restrict__ addterm)
{
    int t = blockIdx.x * 256 + threadIdx.x;            // 65536 = (B*L)*NH
    int bl = t >> 4, h = t & 15;
    const float* xr = x + (size_t)bl * 1024;
    float acc = bg[h];
    for (int k = 0; k < 1024; ++k) acc += xr[k] * Wg[k * 16 + h];
    float gt = 1.0f / (1.0f + __builtin_amdgcn_exp2f(-acc * LOG2E));
    addterm[t] = ((float)err[bl] * gt + amask[bl] + diag[h]) * LOG2E;
}

// vt[(b*16+h)][hd][l] = v[b,l,h,hd]; two l's per thread -> u32 store
__global__ __launch_bounds__(256) void vt_kernel(
    const u16* __restrict__ qkv, u16* __restrict__ vt)
{
    int i = (blockIdx.x * 256 + threadIdx.x) * 2;      // 2M pairs
    int bh = i >> 17;                                  // 64*2048 = 2^17
    int r  = i & 131071;
    int hd = r >> 11, l = r & 2047;                    // l even
    int b = bh >> 4, h = bh & 15;
    const u16* src = qkv + (size_t)(b * 2048 + l) * 3072 + 2048 + h * 64 + hd;
    u32 w = (u32)src[0] | ((u32)src[3072] << 16);
    *(u32*)(vt + i) = w;
}

// ---------------------------------------------------------------------------
// WMMA GEMM: C(M,N) = A(M,K)bf16 x Bt(N,K)bf16^T + bias
// 128x128 block tile, 8 waves (2x4), each wave 64x32 (4x2 of 16x16), K-step 32.
// Double-buffered LDS fed by global_load_async_to_lds_b128 (ASYNCcnt).
// MODE 0: bf16 out, 3-segment bias (qkv).  MODE 1: fp32 out, single bias.
// ---------------------------------------------------------------------------
template <int MODE>
__global__ __launch_bounds__(256) void gemm_kernel(
    const u16* __restrict__ A, const u16* __restrict__ Bt,
    const float* __restrict__ bias0, const float* __restrict__ bias1,
    const float* __restrict__ bias2,
    u16* __restrict__ outb, float* __restrict__ outf,
    int K, int ldc)
{
    __shared__ u16 lA[2][128 * 40];   // padded row stride 40 (80B, 16B-aligned)
    __shared__ u16 lB[2][128 * 40];

    const int t = threadIdx.x;
    const int lane = t & 31, wave = t >> 5;
    const int wm = wave >> 2, wn = wave & 3;
    const int m0 = blockIdx.y * 128, n0 = blockIdx.x * 128;
    const int trow = t >> 1, thalf = (t & 1) * 16;
    const int lane_m = lane & 15, lane_g = lane >> 4;

    v8f acc[4][2] = {};

    // issue one 128x32 A tile + 128x32 B tile into LDS buffer `buf` (async)
    auto issue_tile = [&](int kc, int buf) {
        const u16* ga = A  + (size_t)(m0 + trow) * K + kc + thalf;
        const u16* gb = Bt + (size_t)(n0 + trow) * K + kc + thalf;
        u32 la = (u32)(size_t)&lA[buf][trow * 40 + thalf];
        u32 lb = (u32)(size_t)&lB[buf][trow * 40 + thalf];
        async_b128(la, ga);
        async_b128_o16(la, ga);   // second 16B: +16 on both LDS and global
        async_b128(lb, gb);
        async_b128_o16(lb, gb);
    };

    const int nIter = K >> 5;
    issue_tile(0, 0);

    for (int i = 0; i < nIter; ++i) {
        wait_async0();            // tile i resident in this wave's LDS writes
        __syncthreads();          // visible to all waves; prev buffer free
        if (i + 1 < nIter) issue_tile((i + 1) << 5, (i + 1) & 1);

        const u16* bufA = lA[i & 1];
        const u16* bufB = lB[i & 1];
        FragAB af[4], bfr[2];
        #pragma unroll
        for (int ii = 0; ii < 4; ++ii) {
            // A frag: lane(g,m): u[0..3] <- K=8g..8g+7 ; u[4..7] <- K=16+8g..+7
            const u16* p = bufA + (wm * 64 + ii * 16 + lane_m) * 40;
            af[ii].q[0] = *(const uint4*)(p + 8 * lane_g);
            af[ii].q[1] = *(const uint4*)(p + 16 + 8 * lane_g);
        }
        #pragma unroll
        for (int j = 0; j < 2; ++j) {
            // B frag: lane(g,n): u[0..3] <- K=8g..8g+7 ; u[4..7] <- K=16+8g..+7
            const u16* p = bufB + (wn * 32 + j * 16 + lane_m) * 40;
            bfr[j].q[0] = *(const uint4*)(p + 8 * lane_g);
            bfr[j].q[1] = *(const uint4*)(p + 16 + 8 * lane_g);
        }
        #pragma unroll
        for (int ii = 0; ii < 4; ++ii)
            #pragma unroll
            for (int j = 0; j < 2; ++j)
                acc[ii][j] = WMMA_BF16(af[ii].v, bfr[j].v, acc[ii][j]);
    }

    // D layout: element (m,n): lane = n + 16*(m>=8), vgpr e = m&7
    const int col0 = n0 + wn * 32 + lane_m;            // j=0 column
    #pragma unroll
    for (int i = 0; i < 4; ++i) {
        #pragma unroll
        for (int e = 0; e < 8; ++e) {
            int row = m0 + wm * 64 + i * 16 + e + 8 * lane_g;
            if (MODE == 0) {
                float b0 = (col0 < 1024) ? bias0[col0]
                         : (col0 < 2048) ? bias1[col0 - 1024]
                                         : bias2[col0 - 2048];
                int col1 = col0 + 16;
                float b1 = (col1 < 1024) ? bias0[col1]
                         : (col1 < 2048) ? bias1[col1 - 1024]
                                         : bias2[col1 - 2048];
                u32 w = cvt_pk_bf16(acc[i][0][e] + b0, acc[i][1][e] + b1);
                u16* orow = outb + (size_t)row * ldc + col0;
                orow[0]  = (u16)w;
                orow[16] = (u16)(w >> 16);
            } else {
                float* orow = outf + (size_t)row * ldc + col0;
                orow[0]  = acc[i][0][e] + bias0[col0];
                orow[16] = acc[i][1][e] + bias0[col0 + 16];
            }
        }
    }
}

// ---------------------------------------------------------------------------
// Flash attention: block = 4 waves, each wave owns a 16-row q tile.
// Streams keys in blocks of 32; S and O stay in WMMA fragments; P round-trips
// through LDS to convert D-layout -> A-layout. Softmax runs in the exp2
// domain (bare v_exp_f32); row max reduced on the VALU pipe (permlane16 +
// v_pk_max_num_bf16); row sums are per-lane partials reduced once at the end.
// ---------------------------------------------------------------------------
__global__ __launch_bounds__(128) void attn_kernel(
    const u16* __restrict__ qkv, const u16* __restrict__ vt,
    const float* __restrict__ addterm, u16* __restrict__ attn)
{
    __shared__ u16 lP[4 * 16 * 48];    // per-wave 16x32 P tile, stride 48 (96B)
    const int lane = threadIdx.x & 31;
    const int wave = threadIdx.x >> 5;
    const int bh = blockIdx.y;                 // 0..31
    const int b = bh >> 4, h = bh & 15;
    const int q0 = blockIdx.x * 64 + wave * 16;
    u16* myP = lP + wave * 16 * 48;

    const int m = lane & 15, g = lane >> 4;    // A-lane row / B-lane col; K-group
    const float SCL = 0.125f * LOG2E;          // 1/sqrt(64), exp2 domain

    // Q fragments for rows q0+m, head h (two 32-wide K chunks over HD=64)
    FragAB aq[2];
    {
        const u16* qrow = qkv + ((size_t)(b * 2048 + q0 + m) * 3072 + h * 64);
        #pragma unroll
        for (int c = 0; c < 2; ++c) {
            aq[c].q[0] = *(const uint4*)(qrow + c * 32 + 8 * g);
            aq[c].q[1] = *(const uint4*)(qrow + c * 32 + 16 + 8 * g);
        }
    }

    v8f o0 = {}, o1 = {}, o2 = {}, o3 = {};
    float mrow[8], lrow[8];
    #pragma unroll
    for (int e = 0; e < 8; ++e) { mrow[e] = -3.0e38f; lrow[e] = 0.0f; }

    const u16* vbase = vt + (size_t)bh * 64 * 2048;

    for (int kb = 0; kb < 2048; kb += 32) {
        // ---- scores: S(16x32) = Q(16x64) @ K^T, two 16x16 D tiles
        const u16* krow0 = qkv + (size_t)(b * 2048 + kb + m) * 3072 + 1024 + h * 64;
        const u16* krow1 = krow0 + (size_t)16 * 3072;
        FragAB k0a, k0b, k1a, k1b;
        k0a.q[0] = *(const uint4*)(krow0 + 8 * g);
        k0a.q[1] = *(const uint4*)(krow0 + 16 + 8 * g);
        k0b.q[0] = *(const uint4*)(krow0 + 32 + 8 * g);
        k0b.q[1] = *(const uint4*)(krow0 + 32 + 16 + 8 * g);
        k1a.q[0] = *(const uint4*)(krow1 + 8 * g);
        k1a.q[1] = *(const uint4*)(krow1 + 16 + 8 * g);
        k1b.q[0] = *(const uint4*)(krow1 + 32 + 8 * g);
        k1b.q[1] = *(const uint4*)(krow1 + 32 + 16 + 8 * g);

        v8f s0 = {}, s1 = {};
        s0 = WMMA_BF16(aq[0].v, k0a.v, s0);
        s0 = WMMA_BF16(aq[1].v, k0b.v, s0);
        s1 = WMMA_BF16(aq[0].v, k1a.v, s1);
        s1 = WMMA_BF16(aq[1].v, k1b.v, s1);

        // ---- scaled scores + additive term (per-column, lane-local)
        float a0 = addterm[(size_t)(b * 2048 + kb + m) * 16 + h];
        float a1 = addterm[(size_t)(b * 2048 + kb + 16 + m) * 16 + h];
        float x0[8], x1[8];
        #pragma unroll
        for (int e = 0; e < 8; ++e) {
            x0[e] = s0[e] * SCL + a0;
            x1[e] = s1[e] * SCL + a1;
        }

        // ---- row max: pack row pairs as bf16 (hw cvt), butterfly on VALU
        float rm[8];
        #pragma unroll
        for (int j = 0; j < 4; ++j) {
            float r0 = fmaxf(x0[2 * j],     x1[2 * j]);
            float r1 = fmaxf(x0[2 * j + 1], x1[2 * j + 1]);
            u32 packed = redmax16_bf2(cvt_pk_bf16(r0, r1));
            rm[2 * j]     = __uint_as_float(packed << 16);          // low bf16
            rm[2 * j + 1] = __uint_as_float(packed & 0xFFFF0000u);  // high bf16
        }

        // ---- online softmax update; per-lane partial sums (no x-lane sum)
        float p0[8], p1[8], alpha[8];
        #pragma unroll
        for (int e = 0; e < 8; ++e) {
            float mn = fmaxf(mrow[e], rm[e]);
            alpha[e] = __builtin_amdgcn_exp2f(mrow[e] - mn);
            p0[e] = __builtin_amdgcn_exp2f(x0[e] - mn);
            p1[e] = __builtin_amdgcn_exp2f(x1[e] - mn);
            mrow[e] = mn;
            lrow[e] = lrow[e] * alpha[e] + p0[e] + p1[e];
        }
        #pragma unroll
        for (int e = 0; e < 8; ++e) {
            o0[e] *= alpha[e]; o1[e] *= alpha[e];
            o2[e] *= alpha[e]; o3[e] *= alpha[e];
        }

        // ---- P tile to LDS (D-layout -> memory), then back as A fragment
        #pragma unroll
        for (int e = 0; e < 8; ++e) {
            u32 w = cvt_pk_bf16(p0[e], p1[e]);     // cols m and 16+m of row
            myP[(e + 8 * g) * 48 + m]      = (u16)w;
            myP[(e + 8 * g) * 48 + 16 + m] = (u16)(w >> 16);
        }
        asm volatile("s_wait_dscnt 0" ::: "memory");   // in-wave LDS RAW fence

        FragAB pf;
        pf.q[0] = *(const uint4*)(myP + m * 48 + 8 * g);
        pf.q[1] = *(const uint4*)(myP + m * 48 + 16 + 8 * g);

        // ---- O += P @ V  (V stored transposed: rows = hd, contiguous keys)
        FragAB vf;
        const u16* vr;
        vr = vbase + (size_t)(0 * 16 + m) * 2048 + kb;
        vf.q[0] = *(const uint4*)(vr + 8 * g); vf.q[1] = *(const uint4*)(vr + 16 + 8 * g);
        o0 = WMMA_BF16(pf.v, vf.v, o0);
        vr = vbase + (size_t)(1 * 16 + m) * 2048 + kb;
        vf.q[0] = *(const uint4*)(vr + 8 * g); vf.q[1] = *(const uint4*)(vr + 16 + 8 * g);
        o1 = WMMA_BF16(pf.v, vf.v, o1);
        vr = vbase + (size_t)(2 * 16 + m) * 2048 + kb;
        vf.q[0] = *(const uint4*)(vr + 8 * g); vf.q[1] = *(const uint4*)(vr + 16 + 8 * g);
        o2 = WMMA_BF16(pf.v, vf.v, o2);
        vr = vbase + (size_t)(3 * 16 + m) * 2048 + kb;
        vf.q[0] = *(const uint4*)(vr + 8 * g); vf.q[1] = *(const uint4*)(vr + 16 + 8 * g);
        o3 = WMMA_BF16(pf.v, vf.v, o3);
    }

    // ---- final cross-lane row-sum, normalize, store bf16 (B,L,H)
    #pragma unroll
    for (int e = 0; e < 8; ++e) {
        float tot = redadd16_f32(lrow[e]);
        float inv = 1.0f / tot;
        int row = q0 + e + 8 * g;
        u16* orow = attn + (size_t)(b * 2048 + row) * 1024 + h * 64;
        u32 w0 = cvt_pk_bf16(o0[e] * inv, o1[e] * inv);
        u32 w1 = cvt_pk_bf16(o2[e] * inv, o3[e] * inv);
        orow[m]      = (u16)w0;
        orow[16 + m] = (u16)(w0 >> 16);
        orow[32 + m] = (u16)w1;
        orow[48 + m] = (u16)(w1 >> 16);
    }
}

// ---------------------------------------------------------------------------
// Launch
// ---------------------------------------------------------------------------
extern "C" void kernel_launch(void* const* d_in, const int* in_sizes, int n_in,
                              void* d_out, int out_size, void* d_ws, size_t ws_size,
                              hipStream_t stream)
{
    const float* x     = (const float*)d_in[0];
    const float* amask = (const float*)d_in[1];
    const int*   err   = (const int*)  d_in[2];
    const float* Wq    = (const float*)d_in[3];
    const float* bq    = (const float*)d_in[4];
    const float* Wk    = (const float*)d_in[5];
    const float* bk    = (const float*)d_in[6];
    const float* Wv    = (const float*)d_in[7];
    const float* bv    = (const float*)d_in[8];
    const float* Wo    = (const float*)d_in[9];
    const float* bo    = (const float*)d_in[10];
    const float* diag  = (const float*)d_in[11];
    const float* Wg    = (const float*)d_in[12];
    const float* bg    = (const float*)d_in[13];
    float* out = (float*)d_out;

    u16* ws = (u16*)d_ws;
    u16* wt_qkv = ws;            ws += 3 * 1024 * 1024;   // (3072,1024) bf16
    u16* wt_o   = ws;            ws += 1 * 1024 * 1024;   // (1024,1024) bf16
    u16* xb     = ws;            ws += 4 * 1024 * 1024;   // (4096,1024) bf16
    u16* qkv    = ws;            ws += 12 * 1024 * 1024;  // (4096,3072) bf16
    u16* vt     = ws;            ws += 4 * 1024 * 1024;   // (32,64,2048) bf16
    u16* attn   = ws;            ws += 4 * 1024 * 1024;   // (4096,1024) bf16
    float* addterm = (float*)ws;                          // (4096,16) f32

    prep_w_kernel<<<8192, 256, 0, stream>>>(Wq, Wk, Wv, Wo, wt_qkv, wt_o);
    cvt_x_kernel<<<8192, 256, 0, stream>>>(x, xb);
    gate_kernel<<<256, 256, 0, stream>>>(x, Wg, bg, err, amask, diag, addterm);
    gemm_kernel<0><<<dim3(24, 32), 256, 0, stream>>>(
        xb, wt_qkv, bq, bk, bv, qkv, nullptr, 1024, 3072);
    vt_kernel<<<8192, 256, 0, stream>>>(qkv, vt);
    attn_kernel<<<dim3(32, 32), 128, 0, stream>>>(qkv, vt, addterm, attn);
    gemm_kernel<1><<<dim3(8, 32), 256, 0, stream>>>(
        attn, wt_o, bo, nullptr, nullptr, nullptr, out, 1024, 1024);
}
